// ImprovedSpatialGNN_48601849921798
// MI455X (gfx1250) — matrix operand order repaired and checked by reading, not compiled, not run on hardware
//
#include <hip/hip_runtime.h>
#include <hip/hip_bf16.h>

typedef __attribute__((ext_vector_type(16))) _Float16 v16h;
typedef __attribute__((ext_vector_type(8)))  _Float16 v8h;
typedef __attribute__((ext_vector_type(8)))  float    v8f;

#define N_NODES 20000
#define N_EDGES 160000
#define EA      (N_EDGES + N_NODES)   // edges + self loops
#define NB      128                   // graphs
#define FD      128                   // feature dim
#define NH      4                     // heads
#define OD      128                   // out dim per head
#define HO      (NH * OD)             // 512
#define NL      3
#define BDIM    32
#define MLPD    256
#define NGRP    10000

// ---- WMMA GEMM: C(MxN) = A(MxK,f16 rowmajor) * Bt(NxK,f16 rowmajor)^T -------------
// One wave computes a 16 x (16*NBLK) strip. REQUIREMENT: (Nc/16) % NBLK == 0 and
// K % 32 == 0 and M % 16 == 0 — callers pick NBLK that divides exactly, so the
// kernel body is completely branch-free (no divergent predication around WMMA,
// no spill-inducing control flow). A fragment is loaded once per k-step and
// reused across NBLK WMMAs (~1.2KB cache traffic per wmma instead of 2KB).
template <int NBLK>
__global__ void wmma_gemm_f16f32(const _Float16* __restrict__ A,
                                 const _Float16* __restrict__ Bt,
                                 float* __restrict__ C,
                                 int M, int Nc, int K) {
  int wave = threadIdx.x >> 5;
  int lane = threadIdx.x & 31;
  int ntn = Nc >> 4;                          // 16-wide N tiles
  int ngn = ntn / NBLK;                       // exact by contract
  int ngroups = (M >> 4) * ngn;
  int g = blockIdx.x * (blockDim.x >> 5) + wave;
  if (g >= ngroups) return;                   // whole wave exits together
  int tm = g / ngn;
  int tn0 = (g - tm * ngn) * NBLK;
  int hs  = lane >> 4;                        // lane-half select
  int l16 = lane & 15;
  const _Float16* arow = A + (size_t)(tm * 16 + l16) * K;
  const _Float16* brow = Bt + (size_t)(tn0 * 16 + l16) * K;
  v8f acc[NBLK];
#pragma unroll
  for (int t = 0; t < NBLK; ++t) acc[t] = (v8f){};
  for (int k0 = 0; k0 < K; k0 += 32) {
    union { v16h v; v8h h[2]; } a;
    // 16-bit A 16x32 layout: elems[0..15]=K(k0+16*hs..), contiguous (matches B)
    a.h[0] = *(const v8h*)(arow + k0 + hs * 16);
    a.h[1] = *(const v8h*)(arow + k0 + hs * 16 + 8);
    v16h b[NBLK];
#pragma unroll
    for (int t = 0; t < NBLK; ++t) {
      // B 32x16 layout: lane half selects K block of 16; contiguous along Bt rows
      b[t] = *(const v16h*)(brow + (size_t)t * 16 * K + k0 + hs * 16);
    }
#pragma unroll
    for (int t = 0; t < NBLK; ++t) {
      acc[t] = __builtin_amdgcn_wmma_f32_16x16x32_f16(false, a.v, false, b[t],
                                                      (short)0, acc[t], false, false);
    }
  }
  float* cbase = C + (size_t)(tm * 16 + hs * 8) * Nc + tn0 * 16 + l16;
#pragma unroll
  for (int t = 0; t < NBLK; ++t) {
#pragma unroll
    for (int r = 0; r < 8; ++r) cbase[(size_t)r * Nc + t * 16] = acc[t][r];
  }
}

// ---------------- helpers ----------------
__global__ void k_f32_to_f16(const float* __restrict__ in, _Float16* __restrict__ out, int n) {
  int i = blockIdx.x * blockDim.x + threadIdx.x;
  if (i < n) out[i] = (_Float16)in[i];
}
__global__ void k_zero(unsigned* __restrict__ p, int n) {
  int i = blockIdx.x * blockDim.x + threadIdx.x;
  if (i < n) p[i] = 0u;
}
// Wt[j*FD + f] = Wg[l][h][f][o], j = h*OD + o  (so Wt is Bt for the GAT GEMM)
__global__ void k_prep_W(const float* __restrict__ Wg, _Float16* __restrict__ Wt, int l) {
  int i = blockIdx.x * blockDim.x + threadIdx.x;
  if (i >= HO * FD) return;
  int j = i >> 7, f = i & 127;
  int h = j >> 7, o = j & 127;
  Wt[i] = (_Float16)Wg[(((size_t)(l * NH + h) * FD) + f) * OD + o];
}
// out[c*rows + r] = (f16) in[r*cols + c]
__global__ void k_transpose_to_f16(const float* __restrict__ in, _Float16* __restrict__ out,
                                   int rows, int cols) {
  int i = blockIdx.x * blockDim.x + threadIdx.x;
  if (i >= rows * cols) return;
  int c = i / rows, r = i - c * rows;
  out[i] = (_Float16)in[(size_t)r * cols + c];
}

__device__ __forceinline__ void edge_sd(const int* __restrict__ ei, int e, int& s, int& d) {
  if (e < N_EDGES) { s = ei[e]; d = ei[N_EDGES + e]; } else { s = d = e - N_EDGES; }
}
__device__ __forceinline__ unsigned f2ord(float f) {
  unsigned u = __float_as_uint(f);
  return (u & 0x80000000u) ? ~u : (u | 0x80000000u);
}
__device__ __forceinline__ float ord2f(unsigned u) {
  return __uint_as_float((u & 0x80000000u) ? (u ^ 0x80000000u) : ~u);
}

// ---------------- GAT pieces ----------------
__global__ void k_esed(const float* __restrict__ xw, const float* __restrict__ a_src,
                       const float* __restrict__ a_dst, float* __restrict__ es,
                       float* __restrict__ ed, int l) {
  int i = blockIdx.x * blockDim.x + threadIdx.x;   // n*NH + h
  if (i >= N_NODES * NH) return;
  int n = i >> 2, h = i & 3;
  const float* xp = xw + (size_t)n * HO + h * OD;
  const float* as = a_src + (size_t)(l * NH + h) * OD;
  const float* ad = a_dst + (size_t)(l * NH + h) * OD;
  float s0 = 0.f, s1 = 0.f;
  for (int o = 0; o < OD; ++o) { float v = xp[o]; s0 += v * as[o]; s1 += v * ad[o]; }
  es[i] = s0; ed[i] = s1;
}
__global__ void k_edge_max(const int* __restrict__ ei, const float* __restrict__ es,
                           const float* __restrict__ ed, float* __restrict__ elr,
                           unsigned* __restrict__ men) {
  int i = blockIdx.x * blockDim.x + threadIdx.x;   // e*NH + h
  if (i >= EA * NH) return;
  int e = i >> 2, h = i & 3;
  int s, d; edge_sd(ei, e, s, d);
  float v = es[s * NH + h] + ed[d * NH + h];
  v = v > 0.f ? v : 0.2f * v;
  elr[i] = v;
  atomicMax(&men[d * NH + h], f2ord(v));
}
__global__ void k_edge_exp(const int* __restrict__ ei, const float* __restrict__ elr,
                           const unsigned* __restrict__ men, float* __restrict__ exb,
                           float* __restrict__ den) {
  int i = blockIdx.x * blockDim.x + threadIdx.x;
  if (i >= EA * NH) return;
  int e = i >> 2, h = i & 3;
  int s, d; edge_sd(ei, e, s, d);
  float ex = __expf(elr[i] - ord2f(men[d * NH + h]));
  exb[i] = ex;
  atomicAdd(&den[d * NH + h], ex);
}
__global__ void k_edge_agg(const int* __restrict__ ei, const float* __restrict__ xw,
                           const float* __restrict__ exb, const float* __restrict__ den,
                           float* __restrict__ agg) {
  long long i = (long long)blockIdx.x * blockDim.x + threadIdx.x;  // e*HO + j
  if (i >= (long long)EA * HO) return;
  int e = (int)(i >> 9);
  int j = (int)(i & 511);
  int h = j >> 7;
  int s, d; edge_sd(ei, e, s, d);
  float alpha = exb[e * NH + h] / den[d * NH + h];
  atomicAdd(&agg[(size_t)d * HO + j], xw[(size_t)s * HO + j] * alpha);
}
__global__ void k_headmean(const float* __restrict__ agg, const float* __restrict__ gb,
                           float* __restrict__ hn, int l) {
  int i = blockIdx.x * blockDim.x + threadIdx.x;   // n*OD + o
  if (i >= N_NODES * OD) return;
  int n = i >> 7, o = i & 127;
  const float* ap = agg + (size_t)n * HO + o;
  hn[i] = 0.25f * (ap[0] + ap[OD] + ap[2 * OD] + ap[3 * OD]) + gb[l * OD + o];
}

// ---------------- batchnorm ----------------
__global__ void k_bn_stats(const float* __restrict__ X, float* __restrict__ mean,
                           float* __restrict__ rstd, int rows, int cols) {
  int c = blockIdx.x;
  __shared__ float s1[256], s2[256];
  float a = 0.f, b = 0.f;
  for (int r = threadIdx.x; r < rows; r += blockDim.x) {
    float v = X[(size_t)r * cols + c];
    a += v; b += v * v;
  }
  s1[threadIdx.x] = a; s2[threadIdx.x] = b;
  __syncthreads();
  for (int st = blockDim.x >> 1; st > 0; st >>= 1) {
    if ((int)threadIdx.x < st) { s1[threadIdx.x] += s1[threadIdx.x + st];
                                 s2[threadIdx.x] += s2[threadIdx.x + st]; }
    __syncthreads();
  }
  if (threadIdx.x == 0) {
    float mu = s1[0] / rows;
    float var = s2[0] / rows - mu * mu;
    mean[c] = mu;
    rstd[c] = rsqrtf(var + 1e-5f);
  }
}
__global__ void k_bn_relu_res(const float* __restrict__ hn, const float* __restrict__ mean,
                              const float* __restrict__ rstd, const float* __restrict__ scale,
                              const float* __restrict__ bias, float* __restrict__ hbuf,
                              int l, int useres) {
  int i = blockIdx.x * blockDim.x + threadIdx.x;
  if (i >= N_NODES * OD) return;
  int o = i & 127;
  float v = (hn[i] - mean[o]) * rstd[o] * scale[l * OD + o] + bias[l * OD + o];
  v = v > 0.f ? v : 0.f;
  hbuf[i] = useres ? hbuf[i] + v : v;
}
__global__ void k_bn_relu(float* __restrict__ X, const float* __restrict__ mean,
                          const float* __restrict__ rstd, const float* __restrict__ scale,
                          const float* __restrict__ bias, int rows, int cols) {
  int i = blockIdx.x * blockDim.x + threadIdx.x;
  if (i >= rows * cols) return;
  int c = i % cols;
  float v = (X[i] - mean[c]) * rstd[c] * scale[c] + bias[c];
  X[i] = v > 0.f ? v : 0.f;
}

// ---------------- attention pooling ----------------
__global__ void k_scores(const float* __restrict__ t1, const float* __restrict__ attb1,
                         const float* __restrict__ attA2, const float* __restrict__ attb2,
                         float* __restrict__ s) {
  int n = blockIdx.x * blockDim.x + threadIdx.x;
  if (n >= N_NODES) return;
  const float* tp = t1 + (size_t)n * MLPD;
  float acc = attb2[0];
  for (int k = 0; k < MLPD; ++k) acc += tanhf(tp[k] + attb1[k]) * attA2[k];
  s[n] = acc;
}
__global__ void k_pool_max(const float* __restrict__ s, const int* __restrict__ bidx,
                           unsigned* __restrict__ sm) {
  int n = blockIdx.x * blockDim.x + threadIdx.x;
  if (n >= N_NODES) return;
  atomicMax(&sm[bidx[n]], f2ord(s[n]));
}
__global__ void k_pool_exp(const float* __restrict__ s, const int* __restrict__ bidx,
                           const unsigned* __restrict__ sm, float* __restrict__ se,
                           float* __restrict__ sden) {
  int n = blockIdx.x * blockDim.x + threadIdx.x;
  if (n >= N_NODES) return;
  float e = __expf(s[n] - ord2f(sm[bidx[n]]));
  se[n] = e;
  atomicAdd(&sden[bidx[n]], e);
}
__global__ void k_pool_agg(const float* __restrict__ hbuf, const int* __restrict__ bidx,
                           const float* __restrict__ se, const float* __restrict__ sden,
                           float* __restrict__ ge) {
  int i = blockIdx.x * blockDim.x + threadIdx.x;   // n*OD + o
  if (i >= N_NODES * OD) return;
  int n = i >> 7, o = i & 127;
  int b = bidx[n];
  atomicAdd(&ge[b * OD + o], hbuf[i] * (se[n] / sden[b]));
}
__global__ void k_combined(const float* __restrict__ ge, const float* __restrict__ Eb0,
                           const float* __restrict__ Eb1, const int* __restrict__ b0,
                           const int* __restrict__ b1, float* __restrict__ comb) {
  int i = blockIdx.x * blockDim.x + threadIdx.x;   // b*192 + j
  if (i >= NB * (OD + 2 * BDIM)) return;
  int b = i / (OD + 2 * BDIM), j = i - b * (OD + 2 * BDIM);
  float v;
  if (j < OD) v = ge[b * OD + j];
  else if (j < OD + BDIM) v = Eb0[(size_t)b0[b] * BDIM + (j - OD)];
  else v = Eb1[(size_t)b1[b] * BDIM + (j - OD - BDIM)];
  comb[i] = v;
}

// ---------------- small fp32 FC + finals ----------------
__global__ void k_fc(const float* __restrict__ X, const float* __restrict__ W,
                     const float* __restrict__ bias, float* __restrict__ Y,
                     int rows, int K, int cols) {
  int i = blockIdx.x * blockDim.x + threadIdx.x;
  if (i >= rows * cols) return;
  int r = i / cols, c = i - r * cols;
  const float* xp = X + (size_t)r * K;
  float acc = bias[c];
  for (int k = 0; k < K; ++k) acc += xp[k] * W[(size_t)k * cols + c];
  Y[i] = acc;
}
__global__ void k_gmean(const float* __restrict__ Gemb, float* __restrict__ gm) {
  int j = blockIdx.x * blockDim.x + threadIdx.x;
  if (j >= NGRP) return;
  const float* gp = Gemb + (size_t)j * 64;
  float a = 0.f;
  for (int k = 0; k < 64; ++k) a += gp[k];
  gm[j] = a * (1.f / 64.f);
}
__global__ void k_epilogue(float* __restrict__ out, const float* __restrict__ P3b,
                           const float* __restrict__ gm) {
  int i = blockIdx.x * blockDim.x + threadIdx.x;
  if (i >= NB * NGRP) return;
  int j = i % NGRP;
  out[i] += P3b[j] + 0.1f * gm[j];
}

extern "C" void kernel_launch(void* const* d_in, const int* in_sizes, int n_in,
                              void* d_out, int out_size, void* d_ws, size_t ws_size,
                              hipStream_t stream) {
  const float* x        = (const float*)d_in[0];
  const int*   ei       = (const int*)d_in[1];
  const int*   bidx     = (const int*)d_in[2];
  const int*   b0       = (const int*)d_in[3];
  const int*   b1       = (const int*)d_in[4];
  const float* Wg       = (const float*)d_in[5];
  const float* att_src  = (const float*)d_in[6];
  const float* att_dst  = (const float*)d_in[7];
  const float* gat_bias = (const float*)d_in[8];
  const float* bn_scale = (const float*)d_in[9];
  const float* bn_bias  = (const float*)d_in[10];
  const float* attA1    = (const float*)d_in[11];
  const float* attb1    = (const float*)d_in[12];
  const float* attA2    = (const float*)d_in[13];
  const float* attb2    = (const float*)d_in[14];
  const float* Eb0      = (const float*)d_in[15];
  const float* Eb1      = (const float*)d_in[16];
  const float* P1w      = (const float*)d_in[17];
  const float* P1b      = (const float*)d_in[18];
  const float* pbn1_s   = (const float*)d_in[19];
  const float* pbn1_b   = (const float*)d_in[20];
  const float* P2w      = (const float*)d_in[21];
  const float* P2b      = (const float*)d_in[22];
  const float* pbn2_s   = (const float*)d_in[23];
  const float* pbn2_b   = (const float*)d_in[24];
  const float* P3w      = (const float*)d_in[25];
  const float* P3b      = (const float*)d_in[26];
  const float* Gemb     = (const float*)d_in[27];
  float* out = (float*)d_out;

  char* ws = (char*)d_ws;
  size_t off = 0;
  auto carve = [&](size_t bytes) -> char* {
    char* p = ws + off;
    off = (off + bytes + 255) & ~(size_t)255;
    return p;
  };
  float*     hbuf   = (float*)carve((size_t)N_NODES * FD * 4);
  float*     hn     = (float*)carve((size_t)N_NODES * OD * 4);
  _Float16*  hf16   = (_Float16*)carve((size_t)N_NODES * FD * 2);
  _Float16*  wt     = (_Float16*)carve((size_t)HO * FD * 2);
  float*     xw     = (float*)carve((size_t)N_NODES * HO * 4);   // reused as t1 (N x 256)
  float*     agg    = (float*)carve((size_t)N_NODES * HO * 4);   // reused as P3w^T f16
  float*     es     = (float*)carve((size_t)N_NODES * NH * 4);   // reused as s
  float*     ed     = (float*)carve((size_t)N_NODES * NH * 4);   // reused as se
  unsigned*  men    = (unsigned*)carve((size_t)N_NODES * NH * 4);
  float*     den    = (float*)carve((size_t)N_NODES * NH * 4);
  float*     elr    = (float*)carve((size_t)EA * NH * 4);
  float*     exb    = (float*)carve((size_t)EA * NH * 4);
  float*     mean   = (float*)carve(512 * 4);
  float*     rstd   = (float*)carve(512 * 4);
  unsigned*  sm     = (unsigned*)carve(NB * 4);
  float*     sden   = (float*)carve(NB * 4);
  float*     ge     = (float*)carve((size_t)NB * OD * 4);
  float*     comb   = (float*)carve((size_t)NB * (OD + 2 * BDIM) * 4);
  float*     z1     = (float*)carve((size_t)NB * MLPD * 4);
  float*     z2     = (float*)carve((size_t)NB * 128 * 4);
  _Float16*  z2f16  = (_Float16*)carve((size_t)NB * 128 * 2);
  _Float16*  a1t    = (_Float16*)carve((size_t)MLPD * OD * 2);
  float*     gm     = (float*)carve((size_t)NGRP * 4);
  _Float16*  p3wt   = (_Float16*)agg;   // alias (2.56 MB inside 41 MB)
  float*     t1     = xw;               // alias (N x 256)
  float*     s      = es;               // alias
  float*     se     = ed;               // alias

  auto cdiv = [](long long a, long long b) { return (int)((a + b - 1) / b); };
  const int T = 256;
  // groups = (M/16) * ((N/16)/NBLK), 8 waves per 256-thread block
  auto wmma_blocks = [&](int M, int Nc, int nblk) {
    long long groups = (long long)(M / 16) * ((Nc / 16) / nblk);
    return (int)((groups + 7) / 8);
  };

  // ---------------- 3 GAT layers ----------------
  for (int l = 0; l < NL; ++l) {
    const float* hin = (l == 0) ? x : hbuf;
    k_f32_to_f16<<<cdiv((long long)N_NODES * FD, T), T, 0, stream>>>(hin, hf16, N_NODES * FD);
    k_prep_W<<<cdiv(HO * FD, T), T, 0, stream>>>(Wg, wt, l);
    // xw = h @ Wflat : (20000 x 512), K=128; 32 N-tiles = 8 groups of 4 (exact)
    wmma_gemm_f16f32<4><<<wmma_blocks(N_NODES, HO, 4), T, 0, stream>>>(
        hf16, wt, xw, N_NODES, HO, FD);
    k_esed<<<cdiv(N_NODES * NH, T), T, 0, stream>>>(xw, att_src, att_dst, es, ed, l);
    k_zero<<<cdiv(N_NODES * NH, T), T, 0, stream>>>(men, N_NODES * NH);
    k_zero<<<cdiv(N_NODES * NH, T), T, 0, stream>>>((unsigned*)den, N_NODES * NH);
    k_edge_max<<<cdiv(EA * NH, T), T, 0, stream>>>(ei, es, ed, elr, men);
    k_edge_exp<<<cdiv(EA * NH, T), T, 0, stream>>>(ei, elr, men, exb, den);
    k_zero<<<cdiv((long long)N_NODES * HO, T), T, 0, stream>>>((unsigned*)agg, N_NODES * HO);
    k_edge_agg<<<cdiv((long long)EA * HO, T), T, 0, stream>>>(ei, xw, exb, den, agg);
    k_headmean<<<cdiv(N_NODES * OD, T), T, 0, stream>>>(agg, gat_bias, hn, l);
    k_bn_stats<<<OD, T, 0, stream>>>(hn, mean, rstd, N_NODES, OD);
    k_bn_relu_res<<<cdiv(N_NODES * OD, T), T, 0, stream>>>(
        hn, mean, rstd, bn_scale, bn_bias, hbuf, l, l > 0 ? 1 : 0);
  }

  // ---------------- attention pooling ----------------
  k_f32_to_f16<<<cdiv((long long)N_NODES * FD, T), T, 0, stream>>>(hbuf, hf16, N_NODES * FD);
  k_transpose_to_f16<<<cdiv(OD * MLPD, T), T, 0, stream>>>(attA1, a1t, OD, MLPD);
  // 16 N-tiles = 4 groups of 4 (exact)
  wmma_gemm_f16f32<4><<<wmma_blocks(N_NODES, MLPD, 4), T, 0, stream>>>(
      hf16, a1t, t1, N_NODES, MLPD, FD);
  k_scores<<<cdiv(N_NODES, T), T, 0, stream>>>(t1, attb1, attA2, attb2, s);
  k_zero<<<1, T, 0, stream>>>(sm, NB);
  k_zero<<<1, T, 0, stream>>>((unsigned*)sden, NB);
  k_pool_max<<<cdiv(N_NODES, T), T, 0, stream>>>(s, bidx, sm);
  k_pool_exp<<<cdiv(N_NODES, T), T, 0, stream>>>(s, bidx, sm, se, sden);
  k_zero<<<cdiv(NB * OD, T), T, 0, stream>>>((unsigned*)ge, NB * OD);
  k_pool_agg<<<cdiv(N_NODES * OD, T), T, 0, stream>>>(hbuf, bidx, se, sden, ge);

  // ---------------- head MLP ----------------
  k_combined<<<cdiv(NB * (OD + 2 * BDIM), T), T, 0, stream>>>(ge, Eb0, Eb1, b0, b1, comb);
  k_fc<<<cdiv(NB * MLPD, T), T, 0, stream>>>(comb, P1w, P1b, z1, NB, OD + 2 * BDIM, MLPD);
  k_bn_stats<<<MLPD, T, 0, stream>>>(z1, mean, rstd, NB, MLPD);
  k_bn_relu<<<cdiv(NB * MLPD, T), T, 0, stream>>>(z1, mean, rstd, pbn1_s, pbn1_b, NB, MLPD);
  k_fc<<<cdiv(NB * 128, T), T, 0, stream>>>(z1, P2w, P2b, z2, NB, MLPD, 128);
  k_bn_stats<<<128, T, 0, stream>>>(z2, mean, rstd, NB, 128);
  k_bn_relu<<<cdiv(NB * 128, T), T, 0, stream>>>(z2, mean, rstd, pbn2_s, pbn2_b, NB, 128);

  // ---------------- logits: z2(128x128) @ P3w(128x10000) via WMMA ----------------
  k_f32_to_f16<<<cdiv(NB * 128, T), T, 0, stream>>>(z2, z2f16, NB * 128);
  k_transpose_to_f16<<<cdiv(128 * NGRP, T), T, 0, stream>>>(P3w, p3wt, 128, NGRP);
  k_gmean<<<cdiv(NGRP, T), T, 0, stream>>>(Gemb, gm);
  // 625 N-tiles = 125 groups of 5 (exact)
  wmma_gemm_f16f32<5><<<wmma_blocks(NB, NGRP, 5), T, 0, stream>>>(
      z2f16, p3wt, out, NB, NGRP, 128);
  k_epilogue<<<cdiv(NB * NGRP, T), T, 0, stream>>>(out, P3b, gm);
}
